// PQMatcher_24146306138771
// MI455X (gfx1250) — matrix operand order repaired and checked by reading, not compiled, not run on hardware
//
#include <hip/hip_runtime.h>
#include <hip/hip_bf16.h>
#include <cstdint>
#include <cstddef>

typedef __bf16 bf16_t;
typedef __attribute__((ext_vector_type(16))) __bf16 v16bf;
typedef __attribute__((ext_vector_type(8)))  __bf16 v8bf;
typedef __attribute__((ext_vector_type(8)))  float  v8f;

namespace {
constexpr int kLP = 128, kLQ = 128, kB = 64, kD0 = 150, kH = 150;
constexpr int kHp  = 160;   // padded H / D0 (mult of 32)
constexpr int kKg  = 320;   // folded+padded Wg input: [u(150) pad | c(150) pad]
constexpr int kNg  = 608;   // padded Wg output / r' width (19*32)
constexpr int kNga = 480;   // 3 GRU gates x 160
constexpr int kNvh = 640;   // Wv out (160) + W_hh gates (480)
}  // namespace

__device__ __forceinline__ float sigf(float x) { return 1.0f / (1.0f + __expf(-x)); }

// ---------------------------------------------------------------------------
// Generic bf16 WMMA GEMM:  C[M x NP] = A[M x KP] * W[NP x KP]^T  (f32 accum)
// One wave per 16x16 output tile. Layouts per CDNA5 ISA 7.12.2 (wave32).
// ---------------------------------------------------------------------------
template <int KP, int NP>
__global__ void __launch_bounds__(256) gemm_bf16(const bf16_t* __restrict__ A,
                                                 const bf16_t* __restrict__ W,
                                                 float* __restrict__ C,
                                                 int Mtiles) {
  constexpr int Ntiles = NP / 16;
  int wave = (blockIdx.x << 3) + (threadIdx.x >> 5);
  if (wave >= Mtiles * Ntiles) return;  // wave-uniform: EXEC stays all-ones
  int mt = wave / Ntiles, nt = wave % Ntiles;
  int lane = threadIdx.x & 31;
  int half = lane >> 4, l15 = lane & 15;
  const bf16_t* arow = A + (size_t)(mt * 16 + l15) * KP;
  const bf16_t* wrow = W + (size_t)(nt * 16 + l15) * KP;
  v8f acc = {};
#pragma unroll
  for (int kt = 0; kt < KP / 32; ++kt) {
    const bf16_t* ap = arow + kt * 32 + 8 * half;
    v8bf a0 = *(const v8bf*)ap;
    v8bf a1 = *(const v8bf*)(ap + 16);
    const bf16_t* bp = wrow + kt * 32 + 8 * half;
    v8bf b0 = *(const v8bf*)bp;
    v8bf b1 = *(const v8bf*)(bp + 16);
    v16bf av = __builtin_shufflevector(a0, a1, 0,1,2,3,4,5,6,7,8,9,10,11,12,13,14,15);
    v16bf bv = __builtin_shufflevector(b0, b1, 0,1,2,3,4,5,6,7,8,9,10,11,12,13,14,15);
    acc = __builtin_amdgcn_wmma_f32_16x16x32_bf16(false, av, false, bv,
                                                  (short)0, acc, false, false);
  }
  float* crow = C + (size_t)(mt * 16 + 8 * half) * NP + nt * 16 + l15;
#pragma unroll
  for (int r = 0; r < 8; ++r) crow[(size_t)r * NP] = acc[r];
}

// ---------------------------------------------------------------------------
// Wg GEMM (64 x 608 x 320) with fused sigmoid * r epilogue -> r' in bf16.
// A = xin = [u | c] (folded, padded). r[n] maps back into xin.
// ---------------------------------------------------------------------------
__global__ void __launch_bounds__(256) k_gate_gemm(const bf16_t* __restrict__ A,
                                                   const bf16_t* __restrict__ W,
                                                   bf16_t* __restrict__ Rout) {
  constexpr int Ntiles = kNg / 16;  // 38
  int wave = (blockIdx.x << 3) + (threadIdx.x >> 5);
  if (wave >= 4 * Ntiles) return;
  int mt = wave / Ntiles, nt = wave % Ntiles;
  int lane = threadIdx.x & 31;
  int half = lane >> 4, l15 = lane & 15;
  const bf16_t* arow = A + (size_t)(mt * 16 + l15) * kKg;
  const bf16_t* wrow = W + (size_t)(nt * 16 + l15) * kKg;
  v8f acc = {};
#pragma unroll
  for (int kt = 0; kt < kKg / 32; ++kt) {
    const bf16_t* ap = arow + kt * 32 + 8 * half;
    v8bf a0 = *(const v8bf*)ap;
    v8bf a1 = *(const v8bf*)(ap + 16);
    const bf16_t* bp = wrow + kt * 32 + 8 * half;
    v8bf b0 = *(const v8bf*)bp;
    v8bf b1 = *(const v8bf*)(bp + 16);
    v16bf av = __builtin_shufflevector(a0, a1, 0,1,2,3,4,5,6,7,8,9,10,11,12,13,14,15);
    v16bf bv = __builtin_shufflevector(b0, b1, 0,1,2,3,4,5,6,7,8,9,10,11,12,13,14,15);
    acc = __builtin_amdgcn_wmma_f32_16x16x32_bf16(false, av, false, bv,
                                                  (short)0, acc, false, false);
  }
  int n = nt * 16 + l15;
#pragma unroll
  for (int r = 0; r < 8; ++r) {
    int m = mt * 16 + r + 8 * half;  // m == batch index b
    float g = sigf(acc[r]);
    float rv = 0.f;
    if (n < 600) {
      int k = (n < 150) ? n
            : (n < 300) ? (n - 150)
            : (n < 450) ? (160 + n - 300)
                        : (160 + n - 450);
      rv = (float)A[(size_t)m * kKg + k];
    }
    Rout[(size_t)m * kNg + n] = (bf16_t)(g * rv);
  }
}

// ---------------------------------------------------------------------------
// Attention scores: one wave per (b,l): s = sum_h tanh(wup+wuq+vWv) * V
// ---------------------------------------------------------------------------
__global__ void __launch_bounds__(256) k_attn_score(const float* __restrict__ Wup,
                                                    const float* __restrict__ Wuq,
                                                    const float* __restrict__ vgh,
                                                    const float* __restrict__ V,
                                                    float* __restrict__ s, int step) {
  int wid = ((blockIdx.x * blockDim.x) + threadIdx.x) >> 5;
  int lane = threadIdx.x & 31;
  int b = wid >> 7;
  int l = wid & 127;
  const float* wup = Wup + ((size_t)step * kB + b) * kHp;
  const float* wuq = Wuq + ((size_t)l * kB + b) * kHp;
  const float* vw  = vgh + (size_t)b * kNvh;  // vWv is first 160 of the fused GEMM
  const float* Vb  = V + (size_t)b * kH;
  float acc = 0.f;
#pragma unroll
  for (int t = 0; t < 5; ++t) {
    int h = lane + 32 * t;
    if (h < kH) acc += tanhf(wup[h] + wuq[h] + vw[h]) * Vb[h];
  }
#pragma unroll
  for (int off = 16; off > 0; off >>= 1) acc += __shfl_xor(acc, off, 32);
  if (lane == 0) s[b * kLQ + l] = acc;
}

// ---------------------------------------------------------------------------
// Softmax over l (per b) + context c = a . Uq, builds xin = [u | c] in bf16.
// ---------------------------------------------------------------------------
__global__ void __launch_bounds__(128) k_softmax_ctx(const float* __restrict__ s,
                                                     const float* __restrict__ Up,
                                                     const float* __restrict__ Uq,
                                                     bf16_t* __restrict__ xin, int step) {
  __shared__ float ash[kLQ];
  __shared__ float red[kLQ];
  int b = blockIdx.x, tid = threadIdx.x;  // 128 threads
  float sv = s[b * kLQ + tid];
  red[tid] = sv; __syncthreads();
  for (int off = 64; off > 0; off >>= 1) {
    if (tid < off) red[tid] = fmaxf(red[tid], red[tid + off]);
    __syncthreads();
  }
  float mx = red[0]; __syncthreads();
  float e = __expf(sv - mx);
  red[tid] = e; __syncthreads();
  for (int off = 64; off > 0; off >>= 1) {
    if (tid < off) red[tid] += red[tid + off];
    __syncthreads();
  }
  float a = e / red[0];
  ash[tid] = a; __syncthreads();
  for (int k = tid; k < kHp; k += 128) {
    float u = (k < kD0) ? Up[((size_t)step * kB + b) * kD0 + k] : 0.f;
    xin[(size_t)b * kKg + k] = (bf16_t)u;
    float c = 0.f;
    if (k < kD0) {
      for (int l = 0; l < kLQ; ++l)
        c += ash[l] * Uq[((size_t)l * kB + b) * kD0 + k];
    }
    xin[(size_t)b * kKg + kHp + k] = (bf16_t)c;
  }
}

// ---------------------------------------------------------------------------
// GRU elementwise update (PyTorch GRUCell math); writes v, v_bf, and output.
// ---------------------------------------------------------------------------
__global__ void __launch_bounds__(kHp) k_gru(const float* __restrict__ gi,
                                             const float* __restrict__ vgh,
                                             const float* __restrict__ bip,
                                             const float* __restrict__ bhp,
                                             float* __restrict__ v,
                                             bf16_t* __restrict__ v_bf,
                                             float* __restrict__ out, int step) {
  int b = blockIdx.x, h = threadIdx.x;  // 160 threads
  if (h >= kH) return;
  float ir  = gi[(size_t)b * kNga + h]       + bip[h];
  float iz  = gi[(size_t)b * kNga + 160 + h] + bip[160 + h];
  float in_ = gi[(size_t)b * kNga + 320 + h] + bip[320 + h];
  float hr  = vgh[(size_t)b * kNvh + 160 + h] + bhp[h];
  float hz  = vgh[(size_t)b * kNvh + 320 + h] + bhp[160 + h];
  float hn  = vgh[(size_t)b * kNvh + 480 + h] + bhp[320 + h];
  float rr = sigf(ir + hr);
  float zz = sigf(iz + hz);
  float nn = tanhf(in_ + rr * hn);
  float hv = (1.f - zz) * nn + zz * v[(size_t)b * kHp + h];
  v[(size_t)b * kHp + h] = hv;
  v_bf[(size_t)b * kHp + h] = (bf16_t)hv;
  out[((size_t)step * kB + b) * kH + h] = hv;
}

// --------------------------- one-time prep kernels -------------------------
__global__ void k_cvt_pad(const float* __restrict__ src, bf16_t* __restrict__ dst, int K) {
  int r = blockIdx.x, k = threadIdx.x;  // rows x 160
  dst[(size_t)r * kHp + k] = (bf16_t)((k < K) ? src[(size_t)r * K + k] : 0.f);
}

__global__ void k_fold2(const float* __restrict__ src, bf16_t* __restrict__ dst) {
  // src (150 x 300) -> dst (160 x 160): dst[n][k] = src[n][k] + src[n][k+150]
  int n = blockIdx.x, k = threadIdx.x;
  float v = 0.f;
  if (n < kH && k < kD0) v = src[n * 300 + k] + src[n * 300 + 150 + k];
  dst[(size_t)n * kHp + k] = (bf16_t)v;
}

__global__ void k_prep_wvwhh(const float* __restrict__ Wv, const float* __restrict__ Whh,
                             bf16_t* __restrict__ dst) {
  int n = blockIdx.x, k = threadIdx.x;  // 640 x 160
  float v = 0.f;
  if (k < kH) {
    if (n < kHp) {
      if (n < kH) v = Wv[n * kH + k];
    } else {
      int g = (n - kHp) / kHp, h = (n - kHp) % kHp;
      if (h < kH) v = Whh[(g * kH + h) * kH + k];
    }
  }
  dst[(size_t)n * kHp + k] = (bf16_t)v;
}

__global__ void k_prep_wg(const float* __restrict__ Wg, bf16_t* __restrict__ dst) {
  int n = blockIdx.x, k = threadIdx.x;  // 608 x 320
  float v = 0.f;
  if (n < 600) {
    if (k < 150) {
      v = Wg[n * 600 + k] + Wg[n * 600 + 150 + k];           // u columns folded
    } else if (k >= 160 && k < 310) {
      int kk = k - 160;
      v = Wg[n * 600 + 300 + kk] + Wg[n * 600 + 450 + kk];   // c columns folded
    }
  }
  dst[(size_t)n * kKg + k] = (bf16_t)v;
}

__global__ void k_prep_wih(const float* __restrict__ Wih, bf16_t* __restrict__ dst) {
  int n = blockIdx.x, k = threadIdx.x;  // 480 x 608
  int g = n / kHp, h = n % kHp;
  float v = 0.f;
  if (h < kH && k < 600) v = Wih[(g * kH + h) * 600 + k];
  dst[(size_t)n * kNg + k] = (bf16_t)v;
}

__global__ void k_prep_bias(const float* __restrict__ bih, const float* __restrict__ bhh,
                            float* __restrict__ bip, float* __restrict__ bhp) {
  int t = threadIdx.x;  // 480
  int g = t / kHp, h = t % kHp;
  float vi = 0.f, vh = 0.f;
  if (h < kH) { vi = bih[g * kH + h]; vh = bhh[g * kH + h]; }
  bip[t] = vi;
  bhp[t] = vh;
}

__global__ void k_vinit(const float* __restrict__ v0, float* __restrict__ v,
                        bf16_t* __restrict__ v_bf) {
  int b = blockIdx.x, h = threadIdx.x;  // 64 x 160
  float x = (h < kH) ? v0[b * kH + h] : 0.f;
  v[(size_t)b * kHp + h] = x;
  v_bf[(size_t)b * kHp + h] = (bf16_t)x;
}

// ---------------------------------------------------------------------------
extern "C" void kernel_launch(void* const* d_in, const int* in_sizes, int n_in,
                              void* d_out, int out_size, void* d_ws, size_t ws_size,
                              hipStream_t stream) {
  (void)in_sizes; (void)n_in; (void)out_size; (void)ws_size;
  const float* Up  = (const float*)d_in[0];
  const float* Uq  = (const float*)d_in[1];
  const float* Wp  = (const float*)d_in[2];
  const float* Wq  = (const float*)d_in[3];
  const float* Wv  = (const float*)d_in[4];
  const float* Wg  = (const float*)d_in[5];
  const float* V   = (const float*)d_in[6];
  const float* v0  = (const float*)d_in[7];
  const float* Wih = (const float*)d_in[8];
  const float* Whh = (const float*)d_in[9];
  const float* bih = (const float*)d_in[10];
  const float* bhh = (const float*)d_in[11];
  float* out = (float*)d_out;

  char* p = (char*)d_ws;
  auto alloc = [&](size_t n) -> char* {
    char* r = p;
    p += (n + 255) & ~(size_t)255;
    return r;
  };
  bf16_t* Up_bf = (bf16_t*)alloc((size_t)kLP * kB * kHp * 2);
  bf16_t* Uq_bf = (bf16_t*)alloc((size_t)kLQ * kB * kHp * 2);
  bf16_t* Wq_e  = (bf16_t*)alloc((size_t)kHp * kHp * 2);
  bf16_t* Wp_e  = (bf16_t*)alloc((size_t)kHp * kHp * 2);
  bf16_t* Wvh   = (bf16_t*)alloc((size_t)kNvh * kHp * 2);
  bf16_t* Wg_e  = (bf16_t*)alloc((size_t)kNg * kKg * 2);
  bf16_t* Wih_b = (bf16_t*)alloc((size_t)kNga * kNg * 2);
  float*  bip   = (float*)alloc((size_t)kNga * 4);
  float*  bhp   = (float*)alloc((size_t)kNga * 4);
  float*  Wuq_f = (float*)alloc((size_t)kLQ * kB * kHp * 4);
  float*  Wup_f = (float*)alloc((size_t)kLP * kB * kHp * 4);
  float*  vgh   = (float*)alloc((size_t)kB * kNvh * 4);
  float*  v     = (float*)alloc((size_t)kB * kHp * 4);
  bf16_t* v_bf  = (bf16_t*)alloc((size_t)kB * kHp * 2);
  float*  sbuf  = (float*)alloc((size_t)kB * kLQ * 4);
  bf16_t* xin   = (bf16_t*)alloc((size_t)kB * kKg * 2);
  bf16_t* rbf   = (bf16_t*)alloc((size_t)kB * kNg * 2);
  float*  gi    = (float*)alloc((size_t)kB * kNga * 4);

  // ---- one-time prep: fold/pad/convert (fully re-initializes all scratch) --
  k_cvt_pad<<<kLP * kB, kHp, 0, stream>>>(Up, Up_bf, kD0);
  k_cvt_pad<<<kLQ * kB, kHp, 0, stream>>>(Uq, Uq_bf, kD0);
  k_fold2<<<kHp, kHp, 0, stream>>>(Wq, Wq_e);
  k_fold2<<<kHp, kHp, 0, stream>>>(Wp, Wp_e);
  k_prep_wvwhh<<<kNvh, kHp, 0, stream>>>(Wv, Whh, Wvh);
  k_prep_wg<<<kNg, kKg, 0, stream>>>(Wg, Wg_e);
  k_prep_wih<<<kNga, kNg, 0, stream>>>(Wih, Wih_b);
  k_prep_bias<<<1, kNga, 0, stream>>>(bih, bhh, bip, bhp);
  k_vinit<<<kB, kHp, 0, stream>>>(v0, v, v_bf);

  // ---- hoisted loop-invariant GEMMs: Wuq = Uq.Wq_eff^T, Wup = Up.Wp_eff^T --
  {
    int mt = (kLQ * kB) / 16;  // 512 M-tiles
    int waves = mt * (kHp / 16);
    gemm_bf16<kHp, kHp><<<(waves + 7) / 8, 256, 0, stream>>>(Uq_bf, Wq_e, Wuq_f, mt);
    gemm_bf16<kHp, kHp><<<(waves + 7) / 8, 256, 0, stream>>>(Up_bf, Wp_e, Wup_f, mt);
  }

  // ---- sequential scan ----
  for (int i = 0; i < kLP; ++i) {
    // fused v-GEMM: [vWv | gh] = v . [Wv^T | W_hh^T]
    {
      int waves = 4 * (kNvh / 16);
      gemm_bf16<kHp, kNvh><<<(waves + 7) / 8, 256, 0, stream>>>(v_bf, Wvh, vgh, 4);
    }
    k_attn_score<<<(kB * kLQ) / 8, 256, 0, stream>>>(Wup_f, Wuq_f, vgh, V, sbuf, i);
    k_softmax_ctx<<<kB, kLQ, 0, stream>>>(sbuf, Up, Uq, xin, i);
    {
      int waves = 4 * (kNg / 16);
      k_gate_gemm<<<(waves + 7) / 8, 256, 0, stream>>>(xin, Wg_e, rbf);
    }
    {
      int waves = 4 * (kNga / 16);
      gemm_bf16<kNg, kNga><<<(waves + 7) / 8, 256, 0, stream>>>(rbf, Wih_b, gi, 4);
    }
    k_gru<<<kB, kHp, 0, stream>>>(gi, vgh, bip, bhp, v, v_bf, out, i);
  }
}
